// Decoder_19353122635844
// MI455X (gfx1250) — matrix operand order repaired and checked by reading, not compiled, not run on hardware
//
#include <hip/hip_runtime.h>
#include <hip/hip_bf16.h>
#include <math.h>

// ---------------- problem constants ----------------
#define B_   128
#define T_   200
#define D_   584
#define H_   8
#define HS_  73
#define FF_  2336
#define NTOK (B_*T_)        // 25600 tokens, multiple of 64
#define DP   608            // D padded to multiple of 32
#define HSP  96             // head size padded to multiple of 32 (3 WMMA K-steps)
#define QKVC (3*H_*HSP)     // 2304 packed qkv columns

typedef __attribute__((ext_vector_type(16))) __bf16 v16bf;
typedef __attribute__((ext_vector_type(8)))  __bf16 v8bf;
typedef __attribute__((ext_vector_type(8)))  float  v8f;

#if defined(__HIP_DEVICE_COMPILE__) && __has_builtin(__builtin_amdgcn_global_load_async_to_lds_b128)
#define USE_ASYNC_LDS 1
#else
#define USE_ASYNC_LDS 0
#endif

#if USE_ASYNC_LDS
typedef int v4i_vs __attribute__((vector_size(4 * sizeof(int))));
typedef __attribute__((address_space(1))) v4i_vs* gptr128_t;
typedef __attribute__((address_space(3))) v4i_vs* lptr128_t;
#endif

static __device__ __forceinline__ v16bf cat16(v8bf lo, v8bf hi) {
  return __builtin_shufflevector(lo, hi, 0,1,2,3,4,5,6,7,8,9,10,11,12,13,14,15);
}
static __device__ __forceinline__ v8f wmma_bf16(v16bf a, v16bf b, v8f c) {
  return __builtin_amdgcn_wmma_f32_16x16x32_bf16(false, a, false, b, (short)0, c, false, false);
}
// 16-byte global -> LDS copy: async DMA path on CDNA5, fallback = load+store.
static __device__ __forceinline__ void copy16_g2l(const void* g, void* l) {
#if USE_ASYNC_LDS
  __builtin_amdgcn_global_load_async_to_lds_b128((gptr128_t)(g), (lptr128_t)(l), 0, 0);
#else
  *(v8bf*)l = *(const v8bf*)g;
#endif
}
static __device__ __forceinline__ void wait_async_copies() {
#if USE_ASYNC_LDS
#if __has_builtin(__builtin_amdgcn_s_wait_asynccnt)
  __builtin_amdgcn_s_wait_asynccnt(0);
#else
  asm volatile("s_wait_asynccnt 0" ::: "memory");
#endif
#endif
}

// ---------------- weight packing ----------------
// Wq/Wk/Wv [H,D,HS] f32 -> fused [DP, 3*H*HSP] bf16, zero padded.
__global__ __launch_bounds__(256) void pack_qkv_k(const float* __restrict__ Wq,
                                                  const float* __restrict__ Wk,
                                                  const float* __restrict__ Wv,
                                                  __bf16* __restrict__ o) {
  int idx = blockIdx.x * 256 + threadIdx.x;
  const int total = DP * QKVC;
  if (idx >= total) return;
  int d = idx / QKVC, col = idx % QKVC;
  int sel = col / (H_*HSP);
  int rem = col % (H_*HSP);
  int h = rem / HSP, e = rem % HSP;
  float v = 0.f;
  if (d < D_ && e < HS_) {
    const float* W = (sel == 0) ? Wq : (sel == 1) ? Wk : Wv;
    v = W[((size_t)h * D_ + d) * HS_ + e];
  }
  o[idx] = (__bf16)v;
}

// generic [R,C] f32 -> [Rp,Cp] bf16 zero-padded
__global__ __launch_bounds__(256) void pack_mat_k(const float* __restrict__ in, int R, int C,
                                                  __bf16* __restrict__ o, int Rp, int Cp) {
  int idx = blockIdx.x * 256 + threadIdx.x;
  if (idx >= Rp * Cp) return;
  int r = idx / Cp, c = idx % Cp;
  o[idx] = (r < R && c < C) ? (__bf16)in[(size_t)r * C + c] : (__bf16)0.f;
}

// ---------------- layer norm (f32 in, padded bf16 out) ----------------
__global__ __launch_bounds__(256) void ln_k(const float* __restrict__ x, int ldx,
                                            const float* __restrict__ g,
                                            const float* __restrict__ b,
                                            __bf16* __restrict__ o, int ldo) {
  __shared__ float red[256];
  const int tok = blockIdx.x;
  const int t = threadIdx.x;
  const float* xr = x + (size_t)tok * ldx;
  float s = 0.f;
  for (int c = t; c < D_; c += 256) s += xr[c];
  red[t] = s; __syncthreads();
  for (int w = 128; w > 0; w >>= 1) { if (t < w) red[t] += red[t + w]; __syncthreads(); }
  float mean = red[0] / (float)D_;
  __syncthreads();
  float v = 0.f;
  for (int c = t; c < D_; c += 256) { float d = xr[c] - mean; v += d * d; }
  red[t] = v; __syncthreads();
  for (int w = 128; w > 0; w >>= 1) { if (t < w) red[t] += red[t + w]; __syncthreads(); }
  float rstd = rsqrtf(red[0] / (float)D_ + 1e-5f);
  __bf16* orow = o + (size_t)tok * ldo;
  for (int c = t; c < ldo; c += 256)
    orow[c] = (c < D_) ? (__bf16)((xr[c] - mean) * rstd * g[c] + b[c]) : (__bf16)0.f;
}

// ---------------- WMMA GEMM: C = A(bf16 MxK) * B(bf16 KxN) ----------------
// 8 waves (2x4), macro tile 64x128, wave tile 32x32 (4 WMMA accumulators),
// async A-tile staging + transposed B tile in LDS. Fused bias / relu / residual.
#define BM 64
#define BN 128
#define BK 32
__global__ __launch_bounds__(256) void gemm_k(
    const __bf16* __restrict__ A, int lda,
    const __bf16* __restrict__ Bw, int ldb,
    const float* __restrict__ bias,
    const float* __restrict__ resid, int ldr,
    float* __restrict__ Cf, int ldcf, int cfN,
    __bf16* __restrict__ Cb, int ldcb,
    int M, int K, int Np, int nvalid, int relu) {
  __shared__ __align__(16) __bf16 sA[BM][BK];
  __shared__ __align__(32) __bf16 sBt[BN][BK];   // transposed: [n][k]
  const int t = threadIdx.x;
  const int wave = t >> 5, lane = t & 31;
  const int l15 = lane & 15, lhi = lane >> 4;
  const int wm = wave & 1, wn = wave >> 1;       // 2 x 4 wave grid
  const int m0 = blockIdx.y * BM;
  const int n0 = blockIdx.x * BN;
  const int rbase = wm * 32, cbase = wn * 32;

  const v8f z8 = {0.f,0.f,0.f,0.f,0.f,0.f,0.f,0.f};
  v8f acc[2][2];
  acc[0][0] = z8; acc[0][1] = z8; acc[1][0] = z8; acc[1][1] = z8;

  for (int k0 = 0; k0 < K; k0 += BK) {
    // ---- stage A tile (64x32): one async 16B copy per thread ----
    {
      int row = t >> 2, kc = (t & 3) * 8;
      copy16_g2l(A + (size_t)(m0 + row) * lda + k0 + kc, &sA[row][kc]);
      if (k0 + BK < K)  // prefetch next A tile -> global_prefetch_b8
        __builtin_prefetch(A + (size_t)(m0 + row) * lda + k0 + BK + kc, 0, 0);
    }
    // ---- stage B tile (32x128) transposed into LDS, two 16B vectors/thread ----
    #pragma unroll
    for (int it = 0; it < 2; ++it) {
      int v = t * 2 + it;
      int kr = v >> 4, n = (v & 15) * 8;
      v8bf vb;
      #pragma unroll
      for (int j = 0; j < 8; ++j) vb[j] = (__bf16)0.f;
      if (n0 + n < Np)
        vb = *(const v8bf*)(Bw + (size_t)(k0 + kr) * ldb + n0 + n);
      #pragma unroll
      for (int j = 0; j < 8; ++j) sBt[n + j][kr] = vb[j];
    }
    wait_async_copies();
    __syncthreads();

    // ---- fragments (ISA 16-bit A 16x32 / B 32x16 layouts) ----
    v16bf af[2], bf[2];
    #pragma unroll
    for (int ri = 0; ri < 2; ++ri) {
      int r = rbase + ri * 16 + l15;
      v8bf lo = *(const v8bf*)&sA[r][lhi * 8];
      v8bf hi = *(const v8bf*)&sA[r][lhi * 8 + 16];
      af[ri] = cat16(lo, hi);
    }
    #pragma unroll
    for (int ci = 0; ci < 2; ++ci) {
      int c = cbase + ci * 16 + l15;
      bf[ci] = cat16(*(const v8bf*)&sBt[c][lhi * 8], *(const v8bf*)&sBt[c][lhi * 8 + 16]);
    }
    #pragma unroll
    for (int ri = 0; ri < 2; ++ri)
      #pragma unroll
      for (int ci = 0; ci < 2; ++ci)
        acc[ri][ci] = wmma_bf16(af[ri], bf[ci], acc[ri][ci]);
    __syncthreads();
  }

  // ---- epilogue: C layout -> (row = r + lhi*8, col = l15) ----
  #pragma unroll
  for (int ri = 0; ri < 2; ++ri) {
    #pragma unroll
    for (int ci = 0; ci < 2; ++ci) {
      int col = n0 + cbase + ci * 16 + l15;
      if (col >= Np) continue;
      #pragma unroll
      for (int r = 0; r < 8; ++r) {
        int row = m0 + rbase + ri * 16 + r + lhi * 8;
        float v = acc[ri][ci][r];
        if (bias && col < nvalid) v += bias[col];
        if (relu) v = fmaxf(v, 0.f);
        if (resid && col < nvalid) v += resid[(size_t)row * ldr + col];
        if (Cf && col < cfN) Cf[(size_t)row * ldcf + col] = v;
        if (Cb) Cb[(size_t)row * ldcb + col] = (__bf16)v;
      }
    }
  }
}

// ---------------- causal flash attention, one wave per (qtile, h, b) ----------------
// qkv: [NTOK, QKVC] bf16 (q | k | v blocks, each H*HSP cols).
// out: [NTOK, DP] bf16, head h -> cols [h*HS_, h*HS_+HS_).
__global__ __launch_bounds__(32) void attn_k(const __bf16* __restrict__ qkv,
                                             __bf16* __restrict__ out, float scale) {
  __shared__ __align__(16) __bf16 sP[16 * 32];
  __shared__ __align__(16) __bf16 sV[32 * HSP];
  const int lane = threadIdx.x;
  const int l15 = lane & 15, lhi = lane >> 4;
  const int qt = blockIdx.x, h = blockIdx.y, b = blockIdx.z;
  const int q0 = qt * 16;
  const size_t tokBase = (size_t)b * T_;
  const int qoff = h * HSP;
  const int koff = H_ * HSP + h * HSP;
  const int voff = 2 * H_ * HSP + h * HSP;

  // Q fragments for the 16-row tile, all 3 K-steps (HSP = 96)
  v16bf qf[3];
  {
    int qrow = q0 + l15; if (qrow >= T_) qrow = T_ - 1;
    const __bf16* qp = qkv + (tokBase + qrow) * QKVC + qoff;
    #pragma unroll
    for (int ks = 0; ks < 3; ++ks) {
      v8bf lo = *(const v8bf*)(qp + ks * 32 + lhi * 8);
      v8bf hi = *(const v8bf*)(qp + ks * 32 + lhi * 8 + 16);
      qf[ks] = cat16(lo, hi);
    }
  }

  const v8f z8 = {0.f,0.f,0.f,0.f,0.f,0.f,0.f,0.f};
  v8f oacc[6];
  #pragma unroll
  for (int vc = 0; vc < 6; ++vc) oacc[vc] = z8;
  float mrow[8], lrow[8];
  #pragma unroll
  for (int r = 0; r < 8; ++r) { mrow[r] = -1e30f; lrow[r] = 0.f; }

  const int ktmax = (q0 + 16 + 31) / 32;  // causal: keys beyond q0+15 never needed
  for (int kt = 0; kt < ktmax; ++kt) {
    const int j0 = kt * 32;
    // stage V tile [32 keys x HSP] into LDS via async copies (12 x 16B per lane)
    {
      int key = j0 + lane; int kk = (key < T_) ? key : (T_ - 1);
      const __bf16* vp = qkv + (tokBase + kk) * QKVC + voff;
      #pragma unroll
      for (int j = 0; j < HSP / 8; ++j)
        copy16_g2l(vp + j * 8, &sV[lane * HSP + j * 8]);
    }
    // ---- scores S[16 x 32] = Q K^T over 3 WMMA K-steps ----
    v8f s0 = z8, s1 = z8;
    {
      int keyA = j0 + l15;      int tA = (keyA < T_) ? keyA : (T_ - 1);
      int keyB = j0 + 16 + l15; int tB = (keyB < T_) ? keyB : (T_ - 1);
      const __bf16* kpA = qkv + (tokBase + tA) * QKVC + koff + lhi * 8;
      const __bf16* kpB = qkv + (tokBase + tB) * QKVC + koff + lhi * 8;
      #pragma unroll
      for (int ks = 0; ks < 3; ++ks) {
        v16bf kb0 = cat16(*(const v8bf*)(kpA + ks * 32), *(const v8bf*)(kpA + ks * 32 + 16));
        v16bf kb1 = cat16(*(const v8bf*)(kpB + ks * 32), *(const v8bf*)(kpB + ks * 32 + 16));
        s0 = wmma_bf16(qf[ks], kb0, s0);
        s1 = wmma_bf16(qf[ks], kb1, s1);
      }
    }
    // ---- online softmax (per C-layout row r + lhi*8) ----
    #pragma unroll
    for (int r = 0; r < 8; ++r) {
      int qrow = q0 + r + lhi * 8;
      int key0 = j0 + l15, key1 = j0 + 16 + l15;
      float e0 = s0[r] * scale, e1 = s1[r] * scale;
      bool bad = (qrow >= T_);
      if (bad || key0 > qrow || key0 >= T_) e0 = -1e30f;
      if (bad || key1 > qrow || key1 >= T_) e1 = -1e30f;
      float cm = fmaxf(e0, e1);
      cm = fmaxf(cm, __shfl_xor(cm, 1, 32));
      cm = fmaxf(cm, __shfl_xor(cm, 2, 32));
      cm = fmaxf(cm, __shfl_xor(cm, 4, 32));
      cm = fmaxf(cm, __shfl_xor(cm, 8, 32));
      float mnew = fmaxf(mrow[r], cm);
      float corr = __expf(mrow[r] - mnew);
      float p0 = (e0 < -1e29f) ? 0.f : __expf(e0 - mnew);
      float p1 = (e1 < -1e29f) ? 0.f : __expf(e1 - mnew);
      float rs = p0 + p1;
      rs += __shfl_xor(rs, 1, 32);
      rs += __shfl_xor(rs, 2, 32);
      rs += __shfl_xor(rs, 4, 32);
      rs += __shfl_xor(rs, 8, 32);
      lrow[r] = lrow[r] * corr + rs;
      mrow[r] = mnew;
      #pragma unroll
      for (int vc = 0; vc < 6; ++vc) oacc[vc][r] *= corr;
      int prow = r + lhi * 8;
      sP[prow * 32 + l15]      = (__bf16)p0;
      sP[prow * 32 + 16 + l15] = (__bf16)p1;
    }
    wait_async_copies();
    __syncthreads();
    // ---- O += P V : P reloaded in A-fragment layout, V from LDS as B-fragments ----
    v16bf pf;
    {
      v8bf lo = *(const v8bf*)&sP[l15 * 32 + lhi * 8];
      v8bf hi = *(const v8bf*)&sP[l15 * 32 + lhi * 8 + 16];
      pf = cat16(lo, hi);
    }
    #pragma unroll
    for (int vc = 0; vc < 6; ++vc) {
      v16bf vb;
      #pragma unroll
      for (int i = 0; i < 16; ++i)
        vb[i] = sV[(lhi * 8 + i + (i >= 8 ? 8 : 0)) * HSP + vc * 16 + l15];
      oacc[vc] = wmma_bf16(pf, vb, oacc[vc]);
    }
    __syncthreads();
  }
  // ---- store normalized output (cols e < HS_ only) ----
  #pragma unroll
  for (int r = 0; r < 8; ++r) {
    int qrow = q0 + r + lhi * 8;
    if (qrow >= T_) continue;
    float inv = 1.f / fmaxf(lrow[r], 1e-20f);
    #pragma unroll
    for (int vc = 0; vc < 6; ++vc) {
      int e = vc * 16 + l15;
      if (e < HS_)
        out[(tokBase + qrow) * DP + h * HS_ + e] = (__bf16)(oacc[vc][r] * inv);
    }
  }
}

// ---------------- host orchestration ----------------
extern "C" void kernel_launch(void* const* d_in, const int* in_sizes, int n_in,
                              void* d_out, int out_size, void* d_ws, size_t ws_size,
                              hipStream_t stream) {
  (void)in_sizes; (void)n_in; (void)out_size; (void)ws_size;
  const float* x     = (const float*)d_in[0];
  // d_in[1] enc_out: dead input
  const float* Wq_sa = (const float*)d_in[2];
  const float* Wk_sa = (const float*)d_in[3];
  const float* Wv_sa = (const float*)d_in[4];
  const float* Wo_sa = (const float*)d_in[5];
  const float* bo_sa = (const float*)d_in[6];
  const float* Wq_ca = (const float*)d_in[7];
  const float* Wk_ca = (const float*)d_in[8];
  const float* Wv_ca = (const float*)d_in[9];
  const float* Wo_ca = (const float*)d_in[10];
  const float* bo_ca = (const float*)d_in[11];
  const float* W1    = (const float*)d_in[12];
  const float* b1    = (const float*)d_in[13];
  const float* W2    = (const float*)d_in[14];
  const float* b2    = (const float*)d_in[15];
  const float* g1    = (const float*)d_in[16];
  const float* be1   = (const float*)d_in[17];
  const float* g2    = (const float*)d_in[18];
  const float* be2   = (const float*)d_in[19];
  float* out = (float*)d_out;

  // workspace carving (256B aligned)
  char* ws = (char*)d_ws;
  size_t off = 0;
  auto carve = [&](size_t bytes) -> void* {
    void* p = ws + off;
    off += (bytes + 255) & ~(size_t)255;
    return p;
  };
  __bf16* Wqkv_sa_p = (__bf16*)carve((size_t)DP * QKVC * 2);
  __bf16* Wqkv_ca_p = (__bf16*)carve((size_t)DP * QKVC * 2);
  __bf16* Wo_sa_p   = (__bf16*)carve((size_t)DP * DP * 2);
  __bf16* Wo_ca_p   = (__bf16*)carve((size_t)DP * DP * 2);
  __bf16* W1_p      = (__bf16*)carve((size_t)DP * FF_ * 2);
  __bf16* W2_p      = (__bf16*)carve((size_t)FF_ * DP * 2);
  __bf16* abuf      = (__bf16*)carve((size_t)NTOK * DP * 2);   // h1 / h2
  __bf16* bbuf      = (__bf16*)carve((size_t)NTOK * DP * 2);   // x1 bf16
  __bf16* attnbuf   = (__bf16*)carve((size_t)NTOK * DP * 2);   // concat heads
  __bf16* bigbuf    = (__bf16*)carve((size_t)NTOK * FF_ * 2);  // qkv (2304) / ff1 (2336)
  float*  xf        = (float*) carve((size_t)NTOK * DP * 4);   // running f32 activation

  const float scale = 1.0f / sqrtf((float)D_);

  // --- pack weights ---
  {
    int tq = DP * QKVC;
    pack_qkv_k<<<(tq + 255) / 256, 256, 0, stream>>>(Wq_sa, Wk_sa, Wv_sa, Wqkv_sa_p);
    pack_qkv_k<<<(tq + 255) / 256, 256, 0, stream>>>(Wq_ca, Wk_ca, Wv_ca, Wqkv_ca_p);
    int to = DP * DP;
    pack_mat_k<<<(to + 255) / 256, 256, 0, stream>>>(Wo_sa, D_, D_, Wo_sa_p, DP, DP);
    pack_mat_k<<<(to + 255) / 256, 256, 0, stream>>>(Wo_ca, D_, D_, Wo_ca_p, DP, DP);
    int t1 = DP * FF_;
    pack_mat_k<<<(t1 + 255) / 256, 256, 0, stream>>>(W1, D_, FF_, W1_p, DP, FF_);
    int t2 = FF_ * DP;
    pack_mat_k<<<(t2 + 255) / 256, 256, 0, stream>>>(W2, FF_, D_, W2_p, FF_, DP);
  }

  dim3 blk(256);
  const int Mt = NTOK / BM;  // 400

  // --- 1) h1 = LN(x) ---
  ln_k<<<NTOK, blk, 0, stream>>>(x, D_, g1, be1, abuf, DP);
  // --- 2) qkv_sa = h1 @ Wqkv_sa ---
  gemm_k<<<dim3((QKVC + BN - 1) / BN, Mt), blk, 0, stream>>>(
      abuf, DP, Wqkv_sa_p, QKVC, nullptr, nullptr, 0,
      nullptr, 0, 0, bigbuf, QKVC, NTOK, DP, QKVC, QKVC, 0);
  // --- 3) causal self-attention ---
  attn_k<<<dim3((T_ + 15) / 16, H_, B_), dim3(32), 0, stream>>>(bigbuf, attnbuf, scale);
  // --- 4) x1 = attn @ Wo_sa + bo_sa + x  (f32 -> xf, bf16 -> bbuf) ---
  gemm_k<<<dim3((DP + BN - 1) / BN, Mt), blk, 0, stream>>>(
      attnbuf, DP, Wo_sa_p, DP, bo_sa, x, D_,
      xf, DP, DP, bbuf, DP, NTOK, DP, DP, D_, 0);
  // --- 5) qkv_ca = x1 @ Wqkv_ca (no LN, no residual per source bug) ---
  gemm_k<<<dim3((QKVC + BN - 1) / BN, Mt), blk, 0, stream>>>(
      bbuf, DP, Wqkv_ca_p, QKVC, nullptr, nullptr, 0,
      nullptr, 0, 0, bigbuf, QKVC, NTOK, DP, QKVC, QKVC, 0);
  // --- 6) causal "cross" attention (on x1) ---
  attn_k<<<dim3((T_ + 15) / 16, H_, B_), dim3(32), 0, stream>>>(bigbuf, attnbuf, scale);
  // --- 7) x2 = attn @ Wo_ca + bo_ca (f32 -> xf) ---
  gemm_k<<<dim3((DP + BN - 1) / BN, Mt), blk, 0, stream>>>(
      attnbuf, DP, Wo_ca_p, DP, bo_ca, nullptr, 0,
      xf, DP, DP, nullptr, 0, NTOK, DP, DP, D_, 0);
  // --- 8) h2 = LN(x2) ---
  ln_k<<<NTOK, blk, 0, stream>>>(xf, DP, g2, be2, abuf, DP);
  // --- 9) ff1 = relu(h2 @ W1 + b1) -> bigbuf (stride FF_) ---
  gemm_k<<<dim3((FF_ + BN - 1) / BN, Mt), blk, 0, stream>>>(
      abuf, DP, W1_p, FF_, b1, nullptr, 0,
      nullptr, 0, 0, bigbuf, FF_, NTOK, DP, FF_, FF_, 1);
  // --- 10) out = ff1 @ W2 + b2 + x2 -> d_out (stride D_) ---
  gemm_k<<<dim3((DP + BN - 1) / BN, Mt), blk, 0, stream>>>(
      bigbuf, FF_, W2_p, DP, b2, xf, DP,
      out, D_, D_, nullptr, 0, NTOK, FF_, DP, D_, 0);
}
